// TGN_65506841199023
// MI455X (gfx1250) — matrix operand order repaired and checked by reading, not compiled, not run on hardware
//
#include <hip/hip_runtime.h>
#include <hip/hip_bf16.h>

typedef __attribute__((ext_vector_type(16))) _Float16 v16h;
typedef __attribute__((ext_vector_type(8)))  float    v8f;

#define S_ 32
#define B_ 128
#define N_ 2000
#define L_ 128
#define TILES 125           // N_/16 exactly
#define NCHUNK 8            // node chunks per batch row
#define TPC 16              // tiles per chunk (last chunk: 13)

// ---------------------------------------------------------------------------
// zero workspace (float granularity; sizes below are multiples of 4 bytes)
__global__ void k_zero(float* p, int n) {
    int i = blockIdx.x * blockDim.x + threadIdx.x;
    if (i < n) p[i] = 0.f;
}

// ---------------------------------------------------------------------------
// GRU update for one index array (source or target). One block per batch row.
// Maintains both the f32 memory and its f16 mirror (used by the WMMA kernel).
__global__ __launch_bounds__(384) void k_gru(const int* __restrict__ idx,
                                             const float* __restrict__ x,
                                             const float* __restrict__ t,
                                             const float* __restrict__ tw,
                                             const float* __restrict__ tbv,
                                             const float* __restrict__ W_ih,
                                             const float* __restrict__ W_hh,
                                             const float* __restrict__ b_ih,
                                             const float* __restrict__ b_hh,
                                             float* __restrict__ mem,
                                             _Float16* __restrict__ mem16,
                                             int s) {
    const int b = blockIdx.x, tid = threadIdx.x;
    __shared__ float u[129], h[128], gi[384], gh[384];
    __shared__ int node_s;
    if (tid == 0) node_s = idx[s * B_ + b];
    __syncthreads();
    const int node = node_s;
    const size_t off = ((size_t)(s * B_ + b)) * N_ + node;
    if (tid < 128) {
        u[1 + tid] = __cosf(fmaf(t[off], tw[tid], tbv[tid]));
        h[tid] = mem[node * L_ + tid];
    }
    if (tid == 0) u[0] = x[off];
    __syncthreads();
    float a = b_ih[tid], c = b_hh[tid];
    for (int k = 0; k < 129; ++k) a = fmaf(u[k], W_ih[k * 384 + tid], a);
    for (int k = 0; k < 128; ++k) c = fmaf(h[k], W_hh[k * 384 + tid], c);
    gi[tid] = a; gh[tid] = c;
    __syncthreads();
    if (tid < 128) {
        float r = 1.f / (1.f + __expf(-(gi[tid] + gh[tid])));
        float z = 1.f / (1.f + __expf(-(gi[128 + tid] + gh[128 + tid])));
        float n = tanhf(gi[256 + tid] + r * gh[256 + tid]);
        float hnew = (1.f - z) * n + z * h[tid];
        mem[node * L_ + tid] = hnew;
        mem16[node * L_ + tid] = (_Float16)hnew;
    }
}

// ---------------------------------------------------------------------------
// zt = target_h @ Wt + bt ; also zero the neighbor accumulator
__global__ __launch_bounds__(128) void k_zt(const int* __restrict__ tgt,
                                            const float* __restrict__ x,
                                            const float* __restrict__ tbv,
                                            const float* __restrict__ Wt,
                                            const float* __restrict__ bt,
                                            const float* __restrict__ mem,
                                            float* __restrict__ zt,
                                            float* __restrict__ neigh,
                                            int s) {
    const int b = blockIdx.x, tid = threadIdx.x;
    __shared__ float th[257];
    __shared__ int node_s;
    if (tid == 0) node_s = tgt[s * B_ + b];
    __syncthreads();
    const int node = node_s;
    th[1 + tid] = mem[node * L_ + tid];
    th[129 + tid] = __cosf(tbv[tid]);
    if (tid == 0) th[0] = x[((size_t)(s * B_ + b)) * N_ + node];
    __syncthreads();
    float a = bt[tid];
    for (int k = 0; k < 257; ++k) a = fmaf(th[k], Wt[k * L_ + tid], a);
    zt[b * L_ + tid] = a;
    neigh[b * L_ + tid] = 0.f;
}

// ---------------------------------------------------------------------------
// Big fused kernel: neigh[b,:] += sum_n mask * relu([cos(t*tw+tb) | mem] @
//   [WnT;WnM] + x*Wn0 + bn).  WMMA f16 16x16x32, K = 256.
// mem half of each A tile arrives via async global->LDS DMA (ASYNCcnt),
// overlapped with the VALU cos generation of the dt half.
__global__ __launch_bounds__(256) void k_neigh(const float* __restrict__ x,
                                               const float* __restrict__ t,
                                               const int* __restrict__ mask,
                                               const float* __restrict__ tw,
                                               const float* __restrict__ tbv,
                                               const float* __restrict__ Wn,
                                               const float* __restrict__ bn,
                                               const _Float16* __restrict__ mem16,
                                               float* __restrict__ neigh,
                                               int s) {
    const int b = blockIdx.x;
    const int chunk = blockIdx.y;
    const int tid = threadIdx.x;
    const int lane = tid & 31;
    const int wave = tid >> 5;          // 8 waves, one 16-col tile each
    const int colLocal = lane & 15;
    const int half = lane >> 4;
    const int col = wave * 16 + colLocal;

    __shared__ _Float16 As[16 * 256];   // A tile: 16 nodes x K=256 (8 KB)
    __shared__ float tw_s[128], tb_s[128];
    __shared__ float xv[16], mv[16];

    if (tid < 128) { tw_s[tid] = tw[tid]; tb_s[tid] = tbv[tid]; }

    // B fragments: combined K axis -> rows of Wn.  kk<128: dt part (row 129+kk),
    // kk>=128: mem part (row 1+kk-128).  Held in registers for whole kernel.
    v16h bfr[8];
    #pragma unroll
    for (int kf = 0; kf < 8; ++kf) {
        #pragma unroll
        for (int e = 0; e < 16; ++e) {
            int klocal = 16 * half + e;
            int kk = kf * 32 + klocal;
            int row = (kk < 128) ? (129 + kk) : (1 + (kk - 128));
            bfr[kf][e] = (_Float16)Wn[row * L_ + col];
        }
    }
    const float wn0_col = Wn[col];      // row 0 of Wn (x weight)
    const float bn_col = bn[col];

    const size_t sb = (size_t)(s * B_ + b);
    const float* t_base = t + sb * N_;
    const float* x_base = x + sb * N_;
    const int*   m_base = mask + sb * N_;

    // per-thread fixed slot for the async mem->LDS copy:
    // thread tid moves the 16-byte chunk (row am, halfs ac*8..ac*8+7) of the
    // K=128..255 region:  LDS offset = am*512 + 256 + ac*16 bytes.
    const int am = tid >> 4;            // node row within tile
    const int ac = tid & 15;            // 16B chunk within the 256B mem half
    const unsigned lds_dst =
        (unsigned)(size_t)&As[am * 256 + 128 + ac * 8];   // low 32b = LDS addr

    float colsum = 0.f;
    const int t0 = chunk * TPC;
    const int t1 = (t0 + TPC < TILES) ? (t0 + TPC) : TILES;

    for (int ti = t0; ti < t1; ++ti) {
        const int nbase = ti * 16;
        __syncthreads();                 // previous tile fully consumed

        // --- kick off async DMA of the mem half of A (global f16 -> LDS) ---
        {
            const _Float16* gsrc = mem16 + (size_t)(nbase + am) * L_ + ac * 8;
            asm volatile("global_load_async_to_lds_b128 %0, %1, off"
                         :: "v"(lds_dst), "v"(gsrc) : "memory");
        }

        // --- meanwhile: cos-generate the dt half of A (all 256 threads) ---
        {
            const float tval = t_base[nbase + am];
            #pragma unroll
            for (int i = 0; i < 8; ++i) {
                int kk = ac * 8 + i;
                As[am * 256 + kk] =
                    (_Float16)__cosf(fmaf(tval, tw_s[kk], tb_s[kk]));
            }
            if (tid < 16) {
                xv[tid] = x_base[nbase + tid];
                mv[tid] = (float)m_base[nbase + tid];
            }
        }

        asm volatile("s_wait_asynccnt 0" ::: "memory");
        __syncthreads();

        // --- 8 WMMAs over K=256 ---
        v8f acc = {};
        #pragma unroll
        for (int kf = 0; kf < 8; ++kf) {
            v16h a;
            // A layout: lanes<16 K=0..15; lanes>=16 K=16..31 (matches B)
            const _Float16* p0 = &As[colLocal * 256 + kf * 32 + 16 * half];
            const _Float16* p1 = p0 + 8;
            ((uint4*)&a)[0] = *(const uint4*)p0;
            ((uint4*)&a)[1] = *(const uint4*)p1;
            acc = __builtin_amdgcn_wmma_f32_16x16x32_f16(
                false, a, false, bfr[kf], (short)0, acc, false, false);
        }

        // --- epilogue: rank-1 x term + bias, relu, mask, node reduction ---
        #pragma unroll
        for (int r = 0; r < 8; ++r) {
            const int m = r + 8 * half;  // D layout: VGPR r holds M=r / M=8+r
            float v = acc[r] + xv[m] * wn0_col + bn_col;
            v = v > 0.f ? v : 0.f;
            colsum += v * mv[m];
        }
    }

    // combine the two lane halves (same column) and accumulate globally
    colsum += __shfl_xor(colsum, 16, 32);
    if (half == 0) atomicAdd(&neigh[b * L_ + col], colsum);
}

// ---------------------------------------------------------------------------
// Output MLP: logit = relu([zt|neigh] @ Wo + bo) @ Wl + bl
__global__ __launch_bounds__(128) void k_out(const float* __restrict__ zt,
                                             const float* __restrict__ neigh,
                                             const float* __restrict__ Wo,
                                             const float* __restrict__ bo,
                                             const float* __restrict__ Wl,
                                             const float* __restrict__ bl,
                                             float* __restrict__ out,
                                             int s) {
    const int b = blockIdx.x, tid = threadIdx.x;
    __shared__ float zc[256];
    __shared__ float red[128];
    zc[tid] = zt[b * L_ + tid];
    zc[128 + tid] = neigh[b * L_ + tid];
    __syncthreads();
    float a = bo[tid];
    for (int k = 0; k < 256; ++k) a = fmaf(zc[k], Wo[k * L_ + tid], a);
    a = a > 0.f ? a : 0.f;
    red[tid] = a * Wl[tid];
    __syncthreads();
    for (int st = 64; st > 0; st >>= 1) {
        if (tid < st) red[tid] += red[tid + st];
        __syncthreads();
    }
    if (tid == 0) out[s * B_ + b] = red[0] + bl[0];
}

// ---------------------------------------------------------------------------
extern "C" void kernel_launch(void* const* d_in, const int* in_sizes, int n_in,
                              void* d_out, int out_size, void* d_ws, size_t ws_size,
                              hipStream_t stream) {
    const int*   src  = (const int*)d_in[0];
    const int*   tgt  = (const int*)d_in[1];
    const float* x    = (const float*)d_in[2];
    const float* t    = (const float*)d_in[3];
    const int*   mask = (const int*)d_in[4];
    const float* tw   = (const float*)d_in[5];
    const float* tbv  = (const float*)d_in[6];
    const float* W_ih = (const float*)d_in[7];
    const float* W_hh = (const float*)d_in[8];
    const float* b_ih = (const float*)d_in[9];
    const float* b_hh = (const float*)d_in[10];
    const float* Wn   = (const float*)d_in[11];
    const float* bn   = (const float*)d_in[12];
    const float* Wt   = (const float*)d_in[13];
    const float* bt   = (const float*)d_in[14];
    const float* Wo   = (const float*)d_in[15];
    const float* bo   = (const float*)d_in[16];
    const float* Wl   = (const float*)d_in[17];
    const float* bl   = (const float*)d_in[18];
    float* out = (float*)d_out;

    float*    mem   = (float*)d_ws;                  // N_*L_ f32
    float*    zt    = mem + (size_t)N_ * L_;         // B_*L_ f32
    float*    neigh = zt + (size_t)B_ * L_;          // B_*L_ f32
    _Float16* mem16 = (_Float16*)(neigh + (size_t)B_ * L_);  // N_*L_ f16

    // memory (and its f16 mirror) start at zero
    {
        int n = N_ * L_;
        k_zero<<<(n + 255) / 256, 256, 0, stream>>>(mem, n);
        int n16 = N_ * L_ / 2;   // f16 mirror zeroed as float words
        k_zero<<<(n16 + 255) / 256, 256, 0, stream>>>((float*)mem16, n16);
    }

    for (int s = 0; s < S_; ++s) {
        // src GRU, then tgt GRU (stream order enforces the scatter order)
        k_gru<<<B_, 384, 0, stream>>>(src, x, t, tw, tbv, W_ih, W_hh, b_ih,
                                      b_hh, mem, mem16, s);
        k_gru<<<B_, 384, 0, stream>>>(tgt, x, t, tw, tbv, W_ih, W_hh, b_ih,
                                      b_hh, mem, mem16, s);
        // zt + zero neigh accumulator
        k_zt<<<B_, 128, 0, stream>>>(tgt, x, tbv, Wt, bt, mem, zt, neigh, s);
        // big fused WMMA kernel
        k_neigh<<<dim3(B_, NCHUNK), 256, 0, stream>>>(x, t, mask, tw, tbv, Wn,
                                                      bn, mem16, neigh, s);
        // output MLP
        k_out<<<B_, 128, 0, stream>>>(zt, neigh, Wo, bo, Wl, bl, out, s);
    }
}
